// _ShiftedWindowAttention_29781303230997
// MI455X (gfx1250) — compile-verified, matrix operand order checked by reading, not run on hardware
//
#include <hip/hip_runtime.h>
#include <hip/hip_bf16.h>

// ---------------------------------------------------------------------------
// Shifted-window attention for MI455X (gfx1250), bf16 WMMA pipeline.
//   B=8, L=4096, D=1024, H=8, W=64, hd=128, SHIFT=32
// Three stages, all on v_wmma_f32_16x16x32_bf16 with f32 accumulation:
//   1) qkv = roll(x,-32) @ in_proj_w^T + b         (M=32768,N=3072,K=1024)
//   2) per (window,head): softmax(QK^T/sqrt(hd))V  (64x64x128 tiles)
//   3) out = roll(attn @ out_w^T + b, +32) * 2     (M=32768,N=1024,K=1024)
// Double-buffered LDS staging in both GEMMs; GLOBAL_LOAD_ASYNC_TO_LDS_B128
// (ASYNCcnt-tracked) for all pure-bf16 tile copies.
// ---------------------------------------------------------------------------

typedef __attribute__((ext_vector_type(16))) __bf16 v16bf;
typedef __attribute__((ext_vector_type(8)))  __bf16 v8bf;
typedef __attribute__((ext_vector_type(8)))  float  v8f;
typedef __attribute__((ext_vector_type(4)))  int    v4i;

#define D_MODEL 1024
#define M_TOT   32768          // B*L
#define QKV_STRIDE ((size_t)M_TOT * D_MODEL)   // elements per q/k/v plane

#if __has_builtin(__builtin_amdgcn_global_load_async_to_lds_b128)
#define HAS_ASYNC_LDS 1
typedef __attribute__((address_space(1))) v4i* gptr_v4i;
typedef __attribute__((address_space(3))) v4i* lptr_v4i;
#else
#define HAS_ASYNC_LDS 0
#endif

// 16-byte global -> LDS copy. Async (ASYNCcnt-tracked, no VGPR staging) when
// the toolchain exposes the gfx1250 builtin; plain copy otherwise.
__device__ __forceinline__ void copy_b128_to_lds(void* lds, const void* g) {
#if HAS_ASYNC_LDS
  __builtin_amdgcn_global_load_async_to_lds_b128((gptr_v4i)(g), (lptr_v4i)(lds),
                                                 0, 0);
#else
  *(v8bf*)lds = *(const v8bf*)g;
#endif
}

__device__ __forceinline__ void wait_async_copies() {
#if HAS_ASYNC_LDS
#if __has_builtin(__builtin_amdgcn_s_wait_asynccnt)
  __builtin_amdgcn_s_wait_asynccnt(0);
#else
  asm volatile("s_wait_asynccnt 0x0" ::: "memory");
#endif
#endif
}

__device__ __forceinline__ v8f wmma_bf16(v16bf a, v16bf b, v8f c) {
  // 8 args: (neg_a, A, neg_b, B, c_mod, C, reuse_a, reuse_b)
  return __builtin_amdgcn_wmma_f32_16x16x32_bf16(false, a, false, b,
                                                 (short)0, c, false, false);
}

// A/B fragment load from an LDS tile stored row-major [row][stride] (bf16).
// Matches ISA 16-bit 16x32 A layout: lane l<16 holds row l, K = {k0+0..7,
// k0+16..23}; lane l+16 holds row l, K = {k0+8..15, k0+24..31}.
// Two 16-byte ds loads per fragment (stride*2 and k0*2 must be 16B multiples).
__device__ __forceinline__ v16bf load_frag_row(const __bf16* lds, int row0,
                                               int k0, int stride, int lane) {
  const int l = lane & 15, hi = lane >> 4;
  const __bf16* p = lds + (size_t)(row0 + l) * stride + k0 + hi * 8;
  v8bf lo = *(const v8bf*)(p);
  v8bf hh = *(const v8bf*)(p + 16);
  return __builtin_shufflevector(lo, hh, 0, 1, 2, 3, 4, 5, 6, 7,
                                         8, 9, 10, 11, 12, 13, 14, 15);
}

// B fragment where the logical B matrix is tile[k][col] (e.g. V in P@V):
// lane n holds column col0+n%16, same K packing as above. Scalar gathers.
__device__ __forceinline__ v16bf load_frag_col(const __bf16* lds, int col0,
                                               int k0, int stride, int lane) {
  const int l = lane & 15, hi = lane >> 4;
  const int col = col0 + l;
  const int kb = k0 + hi * 8;
  v16bf r;
#pragma unroll
  for (int e = 0; e < 8; ++e) {
    r[e]     = lds[(size_t)(kb + e) * stride + col];
    r[e + 8] = lds[(size_t)(kb + 16 + e) * stride + col];
  }
  return r;
}

// ---------------------------------------------------------------------------
// Kernel 0: f32 -> bf16 weight conversion (one-time).
// ---------------------------------------------------------------------------
__global__ __launch_bounds__(256) void k_cvt(const float* __restrict__ s,
                                             __bf16* __restrict__ d, int n4) {
  int i = blockIdx.x * 256 + threadIdx.x;
  if (i < n4) {
    float4 f = *(const float4*)(s + (size_t)i * 4);
    size_t o = (size_t)i * 4;
    d[o + 0] = (__bf16)f.x; d[o + 1] = (__bf16)f.y;
    d[o + 2] = (__bf16)f.z; d[o + 3] = (__bf16)f.w;
  }
}

// ---------------------------------------------------------------------------
// Kernel 1: QKV projection GEMM with the -SHIFT roll folded into A gather.
// Block tile 128x128, K-step 32, 8 waves (2M x 4N), wave tile 64x32.
// Double-buffered LDS; B (bf16 weights) staged via async-to-LDS.
// ---------------------------------------------------------------------------
__global__ __launch_bounds__(256) void k_qkv_gemm(
    const float* __restrict__ x, const __bf16* __restrict__ wq,
    const float* __restrict__ bias, __bf16* __restrict__ qkv) {
  __shared__ __bf16 As[2][128][40];  // 80B rows -> 16B-aligned fragments
  __shared__ __bf16 Bs[2][128][40];

  const int tid  = threadIdx.x;
  const int lane = tid & 31, wv = tid >> 5;
  const int wm   = wv >> 2,  wn = wv & 3;
  const int mblk = blockIdx.y << 7, nblk = blockIdx.x << 7;

  const v8f vzero = {0.f, 0.f, 0.f, 0.f, 0.f, 0.f, 0.f, 0.f};
  v8f acc[4][2];
#pragma unroll
  for (int i = 0; i < 4; ++i)
#pragma unroll
    for (int j = 0; j < 2; ++j) acc[i][j] = vzero;

  const int lrow = tid >> 1, lcol = (tid & 1) << 4;
  // cyclic shift by -32 folded into the source row index
  const int gm = mblk + lrow;
  const int bb = gm >> 12, ss = gm & 4095;
  const float*  xrow = x  + ((size_t)(bb << 12) + ((ss + 32) & 4095)) * D_MODEL;
  const __bf16* wrow = wq + (size_t)(nblk + lrow) * D_MODEL;

  // Stage one K-step into LDS buffer `buf`.
  auto stage = [&](int buf, int k0) {
    // A tile: 128x32 f32 -> bf16 (16 values per thread, via VGPR cvt)
    float4 f0 = *(const float4*)(xrow + k0 + lcol);
    float4 f1 = *(const float4*)(xrow + k0 + lcol + 4);
    float4 f2 = *(const float4*)(xrow + k0 + lcol + 8);
    float4 f3 = *(const float4*)(xrow + k0 + lcol + 12);
    __bf16* ap = &As[buf][lrow][lcol];
    ap[0]  = (__bf16)f0.x; ap[1]  = (__bf16)f0.y;
    ap[2]  = (__bf16)f0.z; ap[3]  = (__bf16)f0.w;
    ap[4]  = (__bf16)f1.x; ap[5]  = (__bf16)f1.y;
    ap[6]  = (__bf16)f1.z; ap[7]  = (__bf16)f1.w;
    ap[8]  = (__bf16)f2.x; ap[9]  = (__bf16)f2.y;
    ap[10] = (__bf16)f2.z; ap[11] = (__bf16)f2.w;
    ap[12] = (__bf16)f3.x; ap[13] = (__bf16)f3.y;
    ap[14] = (__bf16)f3.z; ap[15] = (__bf16)f3.w;
    // B tile: bf16 weights, async global->LDS (ASYNCcnt)
    copy_b128_to_lds(&Bs[buf][lrow][lcol],     wrow + k0 + lcol);
    copy_b128_to_lds(&Bs[buf][lrow][lcol + 8], wrow + k0 + lcol + 8);
  };

  stage(0, 0);
  wait_async_copies();
  __syncthreads();

  int buf = 0;
  for (int k0 = 0; k0 < D_MODEL; k0 += 32, buf ^= 1) {
    if (k0 + 32 < D_MODEL) stage(buf ^ 1, k0 + 32);  // prefetch next K-step

    v16bf afr[4], bfr[2];
#pragma unroll
    for (int mt = 0; mt < 4; ++mt)
      afr[mt] = load_frag_row(&As[buf][0][0], wm * 64 + mt * 16, 0, 40, lane);
#pragma unroll
    for (int nt = 0; nt < 2; ++nt)
      bfr[nt] = load_frag_row(&Bs[buf][0][0], wn * 32 + nt * 16, 0, 40, lane);
#pragma unroll
    for (int mt = 0; mt < 4; ++mt)
#pragma unroll
      for (int nt = 0; nt < 2; ++nt)
        acc[mt][nt] = wmma_bf16(afr[mt], bfr[nt], acc[mt][nt]);

    wait_async_copies();
    __syncthreads();
  }

  // Epilogue: +bias, bf16 store into [sel][m][c] planes for the attention pass
  const int l = lane & 15, hi = lane >> 4;
#pragma unroll
  for (int nt = 0; nt < 2; ++nt) {
    const int gcol = nblk + wn * 32 + nt * 16 + l;
    const float bv = bias[gcol];
    const int sel = gcol >> 10, c = gcol & 1023;
    __bf16* dst = qkv + (size_t)sel * QKV_STRIDE;
#pragma unroll
    for (int mt = 0; mt < 4; ++mt)
#pragma unroll
      for (int r = 0; r < 8; ++r) {
        const int gm2 = mblk + wm * 64 + mt * 16 + hi * 8 + r;
        dst[(size_t)gm2 * D_MODEL + c] = (__bf16)(acc[mt][nt][r] + bv);
      }
  }
}

// ---------------------------------------------------------------------------
// Kernel 2: per-(window,head) attention. 4 waves, each owns a 16-row strip.
// Q/K/V staged via async-to-LDS; softmax fully in registers via 16-lane
// shfl_xor reductions (each row lives in one half-wave per the C layout).
// ---------------------------------------------------------------------------
__global__ __launch_bounds__(128) void k_attn(const __bf16* __restrict__ qkv,
                                              __bf16* __restrict__ attn_out) {
  __shared__ __bf16 Qs[64][136];
  __shared__ __bf16 Ks[64][136];
  __shared__ __bf16 Vs[64][136];
  __shared__ __bf16 Pb[64][72];

  const int tid  = threadIdx.x;
  const int lane = tid & 31, wv = tid >> 5;
  const int h = blockIdx.x, win = blockIdx.y;
  const int m0 = win * 64, hc = h * 128;

  // Stage Q/K/V tiles (64x128 bf16 each) into LDS via async copies
  {
    const int row = tid >> 1, half = (tid & 1) * 64;
    const __bf16* qp = qkv + (size_t)(m0 + row) * D_MODEL + hc + half;
#pragma unroll
    for (int i = 0; i < 8; ++i) {
      copy_b128_to_lds(&Qs[row][half + i * 8], qp + i * 8);
      copy_b128_to_lds(&Ks[row][half + i * 8], qp + QKV_STRIDE + i * 8);
      copy_b128_to_lds(&Vs[row][half + i * 8], qp + 2 * QKV_STRIDE + i * 8);
    }
  }
  wait_async_copies();
  __syncthreads();

  // S = Q @ K^T  (wave wv: rows [wv*16, wv*16+16), all 64 cols)
  const v8f vzero = {0.f, 0.f, 0.f, 0.f, 0.f, 0.f, 0.f, 0.f};
  v8f accS[4] = {vzero, vzero, vzero, vzero};
  for (int kc = 0; kc < 128; kc += 32) {
    v16bf a = load_frag_row(&Qs[0][0], wv * 16, kc, 136, lane);
#pragma unroll
    for (int nt = 0; nt < 4; ++nt) {
      v16bf b = load_frag_row(&Ks[0][0], nt * 16, kc, 136, lane);
      accS[nt] = wmma_bf16(a, b, accS[nt]);
    }
  }

  // Softmax per row, entirely in-wave. Row r' = wv*16 + hi*8 + r spans the
  // 16 lanes of one half-wave (cols nt*16 + lane%16), so xor masks 1..8 stay
  // inside the half-wave.
  const int l = lane & 15, hi = lane >> 4;
  const float scale = 0.08838834764831845f;  // 1/sqrt(128)
#pragma unroll
  for (int r = 0; r < 8; ++r) {
    float mx = -3.0e38f;
#pragma unroll
    for (int nt = 0; nt < 4; ++nt) mx = fmaxf(mx, accS[nt][r] * scale);
    for (int off = 1; off < 16; off <<= 1) mx = fmaxf(mx, __shfl_xor(mx, off, 32));
    float ev[4], sum = 0.f;
#pragma unroll
    for (int nt = 0; nt < 4; ++nt) {
      ev[nt] = __expf(accS[nt][r] * scale - mx);
      sum += ev[nt];
    }
    for (int off = 1; off < 16; off <<= 1) sum += __shfl_xor(sum, off, 32);
    const float rinv = 1.0f / sum;
    const int rowi = wv * 16 + hi * 8 + r;
#pragma unroll
    for (int nt = 0; nt < 4; ++nt)
      Pb[rowi][nt * 16 + l] = (__bf16)(ev[nt] * rinv);
  }
  __syncthreads();

  // O = P @ V   (P: 64x64 bf16, V: 64x128) -> 8 N-tiles per wave
  v8f accO[8];
#pragma unroll
  for (int i = 0; i < 8; ++i) accO[i] = vzero;
  for (int kc = 0; kc < 64; kc += 32) {
    v16bf a = load_frag_row(&Pb[0][0], wv * 16, kc, 72, lane);
#pragma unroll
    for (int nt = 0; nt < 8; ++nt) {
      v16bf b = load_frag_col(&Vs[0][0], nt * 16, kc, 136, lane);
      accO[nt] = wmma_bf16(a, b, accO[nt]);
    }
  }
#pragma unroll
  for (int nt = 0; nt < 8; ++nt)
#pragma unroll
    for (int r = 0; r < 8; ++r) {
      const int rowi = wv * 16 + hi * 8 + r;
      attn_out[(size_t)(m0 + rowi) * D_MODEL + hc + nt * 16 + l] =
          (__bf16)accO[nt][r];
    }
}

// ---------------------------------------------------------------------------
// Kernel 3: output projection GEMM + bias + inverse roll (+32) + (o+o).
// Double-buffered; both A (bf16 activations) and B (bf16 weights) staged
// via async-to-LDS.
// ---------------------------------------------------------------------------
__global__ __launch_bounds__(256) void k_out_gemm(
    const __bf16* __restrict__ a, const __bf16* __restrict__ wo,
    const float* __restrict__ bias, float* __restrict__ out) {
  __shared__ __bf16 As[2][128][40];
  __shared__ __bf16 Bs[2][128][40];

  const int tid  = threadIdx.x;
  const int lane = tid & 31, wv = tid >> 5;
  const int wm   = wv >> 2,  wn = wv & 3;
  const int mblk = blockIdx.y << 7, nblk = blockIdx.x << 7;

  const v8f vzero = {0.f, 0.f, 0.f, 0.f, 0.f, 0.f, 0.f, 0.f};
  v8f acc[4][2];
#pragma unroll
  for (int i = 0; i < 4; ++i)
#pragma unroll
    for (int j = 0; j < 2; ++j) acc[i][j] = vzero;

  const int lrow = tid >> 1, lcol = (tid & 1) << 4;
  const __bf16* arow = a  + (size_t)(mblk + lrow) * D_MODEL;
  const __bf16* wrow = wo + (size_t)(nblk + lrow) * D_MODEL;

  auto stage = [&](int buf, int k0) {
    copy_b128_to_lds(&As[buf][lrow][lcol],     arow + k0 + lcol);
    copy_b128_to_lds(&As[buf][lrow][lcol + 8], arow + k0 + lcol + 8);
    copy_b128_to_lds(&Bs[buf][lrow][lcol],     wrow + k0 + lcol);
    copy_b128_to_lds(&Bs[buf][lrow][lcol + 8], wrow + k0 + lcol + 8);
  };

  stage(0, 0);
  wait_async_copies();
  __syncthreads();

  int buf = 0;
  for (int k0 = 0; k0 < D_MODEL; k0 += 32, buf ^= 1) {
    if (k0 + 32 < D_MODEL) stage(buf ^ 1, k0 + 32);

    v16bf afr[4], bfr[2];
#pragma unroll
    for (int mt = 0; mt < 4; ++mt)
      afr[mt] = load_frag_row(&As[buf][0][0], wm * 64 + mt * 16, 0, 40, lane);
#pragma unroll
    for (int nt = 0; nt < 2; ++nt)
      bfr[nt] = load_frag_row(&Bs[buf][0][0], wn * 32 + nt * 16, 0, 40, lane);
#pragma unroll
    for (int mt = 0; mt < 4; ++mt)
#pragma unroll
      for (int nt = 0; nt < 2; ++nt)
        acc[mt][nt] = wmma_bf16(afr[mt], bfr[nt], acc[mt][nt]);

    wait_async_copies();
    __syncthreads();
  }

  const int l = lane & 15, hi = lane >> 4;
#pragma unroll
  for (int nt = 0; nt < 2; ++nt) {
    const int gcol = nblk + wn * 32 + nt * 16 + l;
    const float bv = bias[gcol];
#pragma unroll
    for (int mt = 0; mt < 4; ++mt)
#pragma unroll
      for (int r = 0; r < 8; ++r) {
        const int gm2 = mblk + wm * 64 + mt * 16 + hi * 8 + r;
        const int bb = gm2 >> 12, ss = gm2 & 4095;
        const int orow = (bb << 12) + ((ss + 32) & 4095);  // undo shift
        out[(size_t)orow * D_MODEL + gcol] = 2.0f * (acc[mt][nt][r] + bv);
      }
  }
}

// ---------------------------------------------------------------------------
// Host-side launcher. Inputs: x, in_proj_w, in_proj_b, out_w, out_b (all f32).
// Workspace layout (bf16 elements):
//   wq   [3072*1024]      qkv-projection weights
//   wo   [1024*1024]      output-projection weights
//   qkv  [3*32768*1024]   staged q/k/v
//   ao   [32768*1024]     attention output (pre out-proj)
// Total ~277 MB.
// ---------------------------------------------------------------------------
extern "C" void kernel_launch(void* const* d_in, const int* in_sizes, int n_in,
                              void* d_out, int out_size, void* d_ws,
                              size_t ws_size, hipStream_t stream) {
  const float* x     = (const float*)d_in[0];
  const float* w_in  = (const float*)d_in[1];
  const float* b_in  = (const float*)d_in[2];
  const float* w_out = (const float*)d_in[3];
  const float* b_out = (const float*)d_in[4];
  float* out = (float*)d_out;

  __bf16* wq  = (__bf16*)d_ws;
  __bf16* wo  = wq + (size_t)3072 * 1024;
  __bf16* qkv = wo + (size_t)1024 * 1024;
  __bf16* ao  = qkv + (size_t)3 * M_TOT * D_MODEL;

  // 0) weight conversion
  k_cvt<<<(3072 * 1024 / 4 + 255) / 256, 256, 0, stream>>>(w_in, wq,
                                                           3072 * 1024 / 4);
  k_cvt<<<(1024 * 1024 / 4 + 255) / 256, 256, 0, stream>>>(w_out, wo,
                                                           1024 * 1024 / 4);
  // 1) QKV projection (M=32768, N=3072, K=1024)
  k_qkv_gemm<<<dim3(3072 / 128, M_TOT / 128), 256, 0, stream>>>(x, wq, b_in,
                                                                qkv);
  // 2) windowed attention: grid = (heads, windows)
  k_attn<<<dim3(8, 512), 128, 0, stream>>>(qkv, ao);
  // 3) output projection + roll-back + doubling (M=32768, N=1024, K=1024)
  k_out_gemm<<<dim3(1024 / 128, M_TOT / 128), 256, 0, stream>>>(ao, wo, b_out,
                                                                out);
}
